// KANDense_59777354826115
// MI455X (gfx1250) — matrix operand order repaired and checked by reading, not compiled, not run on hardware
//
#include <hip/hip_runtime.h>

typedef __bf16 v16bf __attribute__((ext_vector_type(16)));
typedef float  v8f   __attribute__((ext_vector_type(8)));

#define B_ROWS 16384
#define D_DIM  1024
#define U_DIM  128
#define TILE_M 128
#define TILE_K 32
#define NHALF  64
#define XS_PAD 36
#define NCHUNK (D_DIM / TILE_K)                 /* 32 */
#define WS_ELEMS ((size_t)NCHUNK * 6 * 8 * 32 * 16)  /* 786432 bf16 = 1.5 MB */
#define LOG2E       1.4426950408889634f
#define NEG5_LOG2E (-7.2134752044448170f)  /* -5 * log2(e) */

// ---- CDNA5 async copy: global -> LDS, 16B per lane, tracked by ASYNCcnt ----
__device__ __forceinline__ void async_cp_b128(const void* gsrc, const void* ldst) {
  // low 32 bits of the generic address of a __shared__ object == LDS byte offset
  unsigned int       loff = (unsigned int)(unsigned long long)(size_t)ldst;
  unsigned long long ga   = (unsigned long long)(size_t)gsrc;
  asm volatile("global_load_async_to_lds_b128 %0, %1, off"
               :: "v"(loff), "v"(ga)
               : "memory");
}
#define WAIT_ASYNCCNT(n) asm volatile("s_wait_asynccnt " #n ::: "memory")

// =====================================================================
// Pre-pass: convert base_weight + 5 spline slices to bf16, laid out in
// exact WMMA B-fragment order: [chunk][mat(6)][ntile(8)][lane(32)][e(16)]
//   n = ntile*16 + (lane&15); k = (lane&16) + e; d = chunk*32 + k
// =====================================================================
__global__ __launch_bounds__(256)
void kan_wprep(const float* __restrict__ wb, const float* __restrict__ ws,
               __bf16* __restrict__ out) {
  int idx = blockIdx.x * 256 + threadIdx.x;           // 0 .. 786431
  int e    = idx & 15;
  int lane = (idx >> 4) & 31;
  int ntg  = (idx >> 9) & 7;
  int cm   = idx >> 12;                               // chunk*6 + mat
  int mat  = cm % 6;
  int c    = cm / 6;
  int n = ntg * 16 + (lane & 15);
  int d = c * TILE_K + (lane & 16) + e;
  float v = (mat == 0) ? wb[(size_t)d * U_DIM + n]
                       : ws[((size_t)d * U_DIM + n) * 8 + (mat - 1)];
  out[idx] = (__bf16)v;
}

// =====================================================================
// Main kernel (async pipelined): double-buffered x tile via
// GLOBAL_LOAD_ASYNC_TO_LDS_B128, weights pre-converted to bf16 fragments.
// =====================================================================
__global__ __launch_bounds__(256)
void kan_dense_wmma_async(const float* __restrict__ x,
                          const __bf16* __restrict__ wf,
                          float* __restrict__ out) {
  __shared__ __attribute__((aligned(16))) float  xs[2][TILE_M][XS_PAD];
  __shared__ __attribute__((aligned(32))) __bf16 bs[6 * 4 * 32 * 16];

  const int t    = threadIdx.x;
  const int wave = t >> 5;
  const int lane = t & 31;
  const int m0   = blockIdx.x * TILE_M;
  const int ny   = blockIdx.y;

  v8f accb[4], accs[4];
#pragma unroll
  for (int i = 0; i < 4; ++i)
#pragma unroll
    for (int r = 0; r < 8; ++r) { accb[i][r] = 0.0f; accs[i][r] = 0.0f; }

  const int xm = t >> 1;            // row of x tile this thread stages
  const int xc = (t & 1) * 16;      // starting column (16 floats = 4x b128)

  // prologue: async-stage chunk 0's x tile into buffer 0
  {
    const float* gx = x + (size_t)(m0 + xm) * D_DIM + xc;
#pragma unroll
    for (int q = 0; q < 4; ++q)
      async_cp_b128(gx + q * 4, &xs[0][xm][xc + q * 4]);
  }

  for (int c = 0; c < NCHUNK; ++c) {
    const int cur = c & 1;

    // async-stage this chunk's 6 weight-fragment tiles (contiguous 16B copies)
#pragma unroll
    for (int mat = 0; mat < 6; ++mat) {
      const __bf16* s = wf + (((size_t)(c * 6 + mat) * 8) + ny * 4) * 512 + t * 8;
      async_cp_b128(s, &bs[mat * 2048 + t * 8]);
    }
    // prefetch next chunk's x tile into the other buffer, then retire all
    // copies except those 4 in-flight prefetches (async loads complete in order)
    if (c + 1 < NCHUNK) {
      const float* gx = x + (size_t)(m0 + xm) * D_DIM + (c + 1) * TILE_K + xc;
#pragma unroll
      for (int q = 0; q < 4; ++q)
        async_cp_b128(gx + q * 4, &xs[cur ^ 1][xm][xc + q * 4]);
      WAIT_ASYNCCNT(4);
    } else {
      WAIT_ASYNCCNT(0);
    }
    __syncthreads();

    // ---- build A fragments in contiguous k order, k = (lane&16) + e, same as B ----
    const int mrow = (wave << 4) + (lane & 15);
    const int koff = (lane & 16);
    float xv[16];
#pragma unroll
    for (int e = 0; e < 16; ++e) {
      int k = koff + e;
      xv[e] = xs[cur][mrow][k];
    }
    v16bf a[6];
#pragma unroll
    for (int e = 0; e < 16; ++e) a[0][e] = (__bf16)xv[e];
#pragma unroll
    for (int g = 0; g < 5; ++g) {
      const float cg = -1.0f + 0.5f * (float)g;
#pragma unroll
      for (int e = 0; e < 16; ++e) {
        float dl = xv[e] - cg;
        a[g + 1][e] = (__bf16)__builtin_amdgcn_exp2f(NEG5_LOG2E * dl * dl);
      }
    }

    // ---- 24 WMMAs: 6 matrices x 4 N-tiles ----
    const v16bf* bp = (const v16bf*)bs;
#pragma unroll
    for (int nt = 0; nt < 4; ++nt) {
      v16bf b0 = bp[nt * 32 + lane];
      accb[nt] = __builtin_amdgcn_wmma_f32_16x16x32_bf16(
          false, a[0], false, b0, (short)0, accb[nt], false, false);
#pragma unroll
      for (int g = 0; g < 5; ++g) {
        v16bf bg = bp[((g + 1) * 4 + nt) * 32 + lane];
        accs[nt] = __builtin_amdgcn_wmma_f32_16x16x32_bf16(
            false, a[g + 1], false, bg, (short)0, accs[nt], false, false);
      }
    }
    __syncthreads();
  }

  // ---- epilogue: out = silu(acc_base) + acc_spline ----
  const int mbase = m0 + (wave << 4) + ((lane & 16) >> 1);
  const int ncol  = ny * NHALF + (lane & 15);
#pragma unroll
  for (int nt = 0; nt < 4; ++nt) {
#pragma unroll
    for (int r = 0; r < 8; ++r) {
      float zb  = accb[nt][r];
      float sig = 1.0f / (1.0f + __builtin_amdgcn_exp2f(-LOG2E * zb));
      out[(size_t)(mbase + r) * U_DIM + ncol + nt * 16] = zb * sig + accs[nt][r];
    }
  }
}

// =====================================================================
// Fallback (round-1 kernel): in-kernel weight conversion, no workspace.
// =====================================================================
__global__ __launch_bounds__(256)
void kan_dense_wmma(const float* __restrict__ x,
                    const float* __restrict__ wb,
                    const float* __restrict__ ws,
                    float* __restrict__ out) {
  __shared__ float xs[TILE_M][XS_PAD];
  __shared__ __attribute__((aligned(32))) __bf16 bs[6 * 4 * 32 * 16];

  const int t    = threadIdx.x;
  const int wave = t >> 5;
  const int lane = t & 31;
  const int m0   = blockIdx.x * TILE_M;
  const int n0   = blockIdx.y * NHALF;

  v8f accb[4], accs[4];
#pragma unroll
  for (int i = 0; i < 4; ++i)
#pragma unroll
    for (int r = 0; r < 8; ++r) { accb[i][r] = 0.0f; accs[i][r] = 0.0f; }

  for (int d0 = 0; d0 < D_DIM; d0 += TILE_K) {
    {
      const int m  = t >> 1;
      const int c0 = (t & 1) * 16;
      const float* gx = x + (size_t)(m0 + m) * D_DIM + d0 + c0;
      float4 v0 = *(const float4*)(gx + 0);
      float4 v1 = *(const float4*)(gx + 4);
      float4 v2 = *(const float4*)(gx + 8);
      float4 v3 = *(const float4*)(gx + 12);
      float* dst = &xs[m][c0];
      *(float4*)(dst + 0)  = v0;
      *(float4*)(dst + 4)  = v1;
      *(float4*)(dst + 8)  = v2;
      *(float4*)(dst + 12) = v3;
      if (d0 + TILE_K < D_DIM)
        __builtin_prefetch(gx + TILE_K, 0, 1);
    }
    {
      const int k  = t >> 3;
      const int nn = (t & 7) * 8;
      const int kb = (k & 16);
      const int e  = k & 15;
      {
        const float* s = wb + (size_t)(d0 + k) * U_DIM + n0 + nn;
        float4 a4 = *(const float4*)(s);
        float4 b4 = *(const float4*)(s + 4);
        float vals[8] = {a4.x, a4.y, a4.z, a4.w, b4.x, b4.y, b4.z, b4.w};
#pragma unroll
        for (int i = 0; i < 8; ++i) {
          int n = nn + i;
          bs[(((n >> 4) * 32 + ((n & 15) + kb)) << 4) + e] = (__bf16)vals[i];
        }
      }
#pragma unroll
      for (int g = 0; g < 5; ++g) {
        const float* s = ws + ((size_t)(d0 + k) * U_DIM + n0 + nn) * 8 + g;
        float vals[8];
#pragma unroll
        for (int i = 0; i < 8; ++i) vals[i] = s[i * 8];
#pragma unroll
        for (int i = 0; i < 8; ++i) {
          int n = nn + i;
          bs[((((g + 1) * 4 + (n >> 4)) * 32 + ((n & 15) + kb)) << 4) + e] =
              (__bf16)vals[i];
        }
      }
    }
    __syncthreads();

    const int mrow = (wave << 4) + (lane & 15);
    const int koff = (lane & 16);
    float xv[16];
#pragma unroll
    for (int e = 0; e < 16; ++e) {
      int k = koff + e;
      xv[e] = xs[mrow][k];
    }
    v16bf a[6];
#pragma unroll
    for (int e = 0; e < 16; ++e) a[0][e] = (__bf16)xv[e];
#pragma unroll
    for (int g = 0; g < 5; ++g) {
      const float cg = -1.0f + 0.5f * (float)g;
#pragma unroll
      for (int e = 0; e < 16; ++e) {
        float dl = xv[e] - cg;
        a[g + 1][e] = (__bf16)__builtin_amdgcn_exp2f(NEG5_LOG2E * dl * dl);
      }
    }

    const v16bf* bp = (const v16bf*)bs;
#pragma unroll
    for (int nt = 0; nt < 4; ++nt) {
      v16bf b0 = bp[nt * 32 + lane];
      accb[nt] = __builtin_amdgcn_wmma_f32_16x16x32_bf16(
          false, a[0], false, b0, (short)0, accb[nt], false, false);
#pragma unroll
      for (int g = 0; g < 5; ++g) {
        v16bf bg = bp[((g + 1) * 4 + nt) * 32 + lane];
        accs[nt] = __builtin_amdgcn_wmma_f32_16x16x32_bf16(
            false, a[g + 1], false, bg, (short)0, accs[nt], false, false);
      }
    }
    __syncthreads();
  }

  const int mbase = m0 + (wave << 4) + ((lane & 16) >> 1);
  const int ncol  = n0 + (lane & 15);
#pragma unroll
  for (int nt = 0; nt < 4; ++nt) {
#pragma unroll
    for (int r = 0; r < 8; ++r) {
      float zb  = accb[nt][r];
      float sig = 1.0f / (1.0f + __builtin_amdgcn_exp2f(-LOG2E * zb));
      out[(size_t)(mbase + r) * U_DIM + ncol + nt * 16] = zb * sig + accs[nt][r];
    }
  }
}

extern "C" void kernel_launch(void* const* d_in, const int* in_sizes, int n_in,
                              void* d_out, int out_size, void* d_ws, size_t ws_size,
                              hipStream_t stream) {
  const float* x  = (const float*)d_in[0];   // [16384, 1024] f32
  const float* wb = (const float*)d_in[1];   // [1024, 128]   f32
  const float* ws = (const float*)d_in[2];   // [1024, 128, 8] f32
  float* out = (float*)d_out;                // [16384, 128]  f32
  dim3 grid(B_ROWS / TILE_M, U_DIM / NHALF, 1);
  if (ws_size >= WS_ELEMS * sizeof(unsigned short)) {
    kan_wprep<<<dim3((unsigned)(WS_ELEMS / 256)), 256, 0, stream>>>(
        wb, ws, (__bf16*)d_ws);
    kan_dense_wmma_async<<<grid, 256, 0, stream>>>(
        x, (const __bf16*)d_ws, out);
  } else {
    kan_dense_wmma<<<grid, 256, 0, stream>>>(x, wb, ws, out);
  }
  (void)in_sizes; (void)n_in; (void)out_size;
}
